// MLP_single_wavelength_att_2877628088459
// MI455X (gfx1250) — matrix operand order repaired and checked by reading, not compiled, not run on hardware
//
#include <hip/hip_runtime.h>

// ---------------------------------------------------------------------------
// MI455X (gfx1250) implementation of the wavelength-attention MLP.
//  - all matmuls are M=16 v_wmma_f32_16x16x32_bf16 tiles
//  - pack kernels pre-swizzle weights into bf16 B-fragment layout with
//    layernorm scale/bias folded in
//  - main kernel: 1 wave32 per workgroup handles 16 batch elements,
//    LDS footprint 100KB/wave -> 3 workgroups per WGP (3/4 SIMDs busy)
//  - branch-free GELU (sigmoid form), A-fragments hoisted into VGPRs
// ---------------------------------------------------------------------------

typedef __attribute__((ext_vector_type(16))) __bf16 v16bf;
typedef __attribute__((ext_vector_type(8)))  __bf16 v8bf;
typedef __attribute__((ext_vector_type(8)))  float  v8f;
typedef __attribute__((ext_vector_type(4)))  float  v4f;

// Problem dims
constexpr int BATCH = 16384;
constexpr int DP    = 32;
constexpr int D     = 128;
constexpr int NL    = 10;

// Fragment geometry: one B fragment = 32 lanes x 16 bf16 = 512 halves = 1KB
constexpr int FH = 512;

// Fragment-region layout (units: fragments)
constexpr int F_WIN1  = 0;                       // 1 kt x 32 nt  = 32
constexpr int F_WIN2  = 32;                      // 16 kt x 128 nt = 2048
constexpr int F_LAYER = 2080;                    // per-layer block
constexpr int F_PER_LAYER = 256;
constexpr int FL_Q = 0, FL_K = 32, FL_V = 64, FL_O = 96, FL_M1 = 128, FL_M2 = 192;
constexpr int F_WO1 = F_LAYER + NL * F_PER_LAYER;      // 4 kt x 16 nt = 64
constexpr int F_WO2 = F_WO1 + 64;                      // 8 kt x 16 nt = 128
constexpr int F_WO3 = F_WO2 + 128;                     // 8 kt x 1 nt = 8 (N padded)
constexpr int F_TOTAL = F_WO3 + 8;                     // 4840 frags
constexpr size_t WBYTES = (size_t)F_TOTAL * FH * 2;    // ~4.96 MB

// Bias region layout (units: f32), appended after fragments in d_ws
constexpr int BI_IN1 = 0;            // 512
constexpr int BI_IN2 = 512;          // 2048
constexpr int BI_LAYER = 2560;       // per layer 896
constexpr int BI_PER_LAYER = 896;
constexpr int BL_Q = 0, BL_K = 128, BL_V = 256, BL_O = 384, BL_M1 = 512, BL_M2 = 768;
constexpr int BI_O1 = BI_LAYER + NL * BI_PER_LAYER;   // 256
constexpr int BI_O2 = BI_O1 + 256;                    // 256
constexpr int BI_O3 = BI_O2 + 256;                    // 2
constexpr int BI_TOTAL = BI_O3 + 16;
constexpr size_t WS_TOTAL = WBYTES + (size_t)BI_TOTAL * 4;

// ---------------------------------------------------------------------------
// LDS layout, 100KB per wave (1 wave per workgroup -> 3 WGs per 320KB WGP).
// Region A (16KB) and region B (12KB) are unions of phase-disjoint buffers.
// ---------------------------------------------------------------------------
constexpr int S_NKV = 0;          // 16 elems x 16 tok x 128 bf16 = 65536 (normalized enc_p)
constexpr int S_XF  = 65536;      // 16 x 128 f32 residual x      = 8192
constexpr int S_A   = 73728;      // 16KB union:
                                  //   encoder: hb [16 x 512 bf16]
                                  //   layers : qf / kf / vf / ob  (4KB each, bf16)
                                  //   head   : ha [16x256 bf16] / hc [16x256 bf16]
constexpr int S_B   = 90112;      // 12KB union:
                                  //   encoder: pb [16 x 32 f32]
                                  //   layers : hh [16x128 bf16] + tb [16x256 bf16]
constexpr int SMEM_BYTES = 102400;

// ---------------------------------------------------------------------------
// device helpers
// ---------------------------------------------------------------------------
__device__ __forceinline__ void lds_fence() {
  asm volatile("s_wait_dscnt 0" ::: "memory");
}

__device__ __forceinline__ v8f v8f_zero() {
  v8f c = {0.f, 0.f, 0.f, 0.f, 0.f, 0.f, 0.f, 0.f};
  return c;
}

// Branch-free tanh-GELU: 0.5*(1+tanh(z)) == sigmoid(2z)
__device__ __forceinline__ float gelu_f(float x) {
  float z = 1.5957691216f * (x + 0.044715f * x * x * x);  // 2*sqrt(2/pi)*(...)
  float s = __builtin_amdgcn_rcpf(1.f + __expf(-z));
  return x * s;
}

__device__ __forceinline__ v8f wmma_bf16(v16bf a, v16bf b, v8f c) {
  return __builtin_amdgcn_wmma_f32_16x16x32_bf16(
      /*neg_a=*/false, a, /*neg_b=*/false, b,
      /*c_mod=*/(short)0, c, /*reuse_a=*/false, /*reuse_b=*/false);
}

// A-fragment (16x32 MxK) from bf16 row-major LDS buffer.
// lane L: row M = L%16; halves 0..7 -> K = kb..kb+7; halves 8..15 -> K+16,
// kb = 8*(L>=16). Two contiguous 16B ds loads.
__device__ __forceinline__ v16bf loadA_bf16(const __bf16* base, int stride, int K0, int lane) {
  int row = lane & 15;
  int kb  = (lane >> 4) << 3;
  const v8bf* pp = (const v8bf*)(base + row * stride + K0 + kb);
  v8bf lo = pp[0];
  v8bf hi = pp[2];  // +16 halves
  v16bf r;
#pragma unroll
  for (int i = 0; i < 8; i++) { r[i] = lo[i]; r[8 + i] = hi[i]; }
  return r;
}

// A-fragment from f32 row-major LDS buffer (convert to bf16 on the fly).
__device__ __forceinline__ v16bf loadA_f32(const float* base, int stride, int K0, int lane) {
  int row = lane & 15;
  int kb  = (lane >> 4) << 3;
  const float* pp = base + row * stride + K0 + kb;
  const v4f* q0 = (const v4f*)pp;
  const v4f* q1 = (const v4f*)(pp + 16);
  v4f a0 = q0[0], a1 = q0[1], b0 = q1[0], b1 = q1[1];
  v16bf r;
#pragma unroll
  for (int i = 0; i < 4; i++) {
    r[i]      = (__bf16)a0[i];
    r[4 + i]  = (__bf16)a1[i];
    r[8 + i]  = (__bf16)b0[i];
    r[12 + i] = (__bf16)b1[i];
  }
  return r;
}

// B-fragment from pre-packed global weights: lane L holds 16 contiguous bf16.
__device__ __forceinline__ v16bf loadB(const unsigned short* wf, int frag, int lane) {
  const v8bf* pp = (const v8bf*)(wf + (size_t)frag * FH + lane * 16);
  v8bf lo = pp[0];
  v8bf hi = pp[1];
  v16bf r;
#pragma unroll
  for (int i = 0; i < 8; i++) { r[i] = lo[i]; r[8 + i] = hi[i]; }
  return r;
}

// ---------------------------------------------------------------------------
// pack kernels
// ---------------------------------------------------------------------------
__global__ __launch_bounds__(32) void pack_w(const float* __restrict__ W,
                                             const float* __restrict__ rowscale,
                                             int K, int N, int NT,
                                             unsigned short* __restrict__ dst_u16) {
  __bf16* dst = (__bf16*)dst_u16;
  int frag = blockIdx.x;
  int lane = threadIdx.x;
  int kt = frag / NT, nt = frag % NT;
  int nn = nt * 16 + (lane & 15);
  int kb = kt * 32 + ((lane >> 4) << 3);
  __bf16* o = dst + (size_t)frag * FH + lane * 16;
#pragma unroll
  for (int i = 0; i < 16; i++) {
    int k = kb + (i < 8 ? i : i + 8);
    float v = 0.f;
    if (k < K && nn < N) {
      v = W[(size_t)k * N + nn];
      if (rowscale) v *= rowscale[k];
    }
    o[i] = (__bf16)v;
  }
}

__global__ void fold_bias(const float* __restrict__ b, const float* __restrict__ lb,
                          const float* __restrict__ W, int K, int N,
                          float* __restrict__ dst) {
  int nn = blockIdx.x * blockDim.x + threadIdx.x;
  if (nn >= N) return;
  float s = b ? b[nn] : 0.f;
  if (lb && W)
    for (int k = 0; k < K; k++) s += lb[k] * W[(size_t)k * N + nn];
  dst[nn] = s;
}

// ---------------------------------------------------------------------------
// main kernel: one wave32 per workgroup handles 16 batch elements
// ---------------------------------------------------------------------------
__global__ __launch_bounds__(32) void mlp_att_kernel(const float* __restrict__ p,
                                                     const float* __restrict__ w,
                                                     const unsigned short* __restrict__ Wf,
                                                     const float* __restrict__ Bf,
                                                     float* __restrict__ out) {
  extern __shared__ char smem[];
  __bf16* nkv = (__bf16*)(smem + S_NKV);
  float*  xf  = (float*)(smem + S_XF);
  // region A aliases
  __bf16* hb  = (__bf16*)(smem + S_A);             // encoder hidden [16x512]
  __bf16* qf  = (__bf16*)(smem + S_A);             // [16x128]
  __bf16* kf  = (__bf16*)(smem + S_A + 4096);      // [16 tok x 128]
  __bf16* vf  = (__bf16*)(smem + S_A + 8192);      // [16 tok x 128]
  __bf16* ob  = (__bf16*)(smem + S_A + 12288);     // [16x128]
  __bf16* ha  = (__bf16*)(smem + S_A);             // head stage 1 [16x256]
  __bf16* hc  = (__bf16*)(smem + S_A + 8192);      // head stage 2 [16x256]
  // region B aliases
  float*  pb  = (float*)(smem + S_B);              // staged p [16x32]
  __bf16* hh  = (__bf16*)(smem + S_B);             // normalized x [16x128]
  __bf16* tb  = (__bf16*)(smem + S_B + 4096);      // mlp hidden [16x256]

  const int lane = threadIdx.x;
  const int eb   = blockIdx.x * 16;        // first batch element of this wave
  const int n    = lane & 15;              // C/D column within tile
  const int rb   = (lane >> 4) << 3;       // C/D row base (0 or 8)

  // ---- stage p rows, build frequency encoding x ----
  for (int t = lane; t < 16 * DP; t += 32) pb[t] = p[(size_t)eb * DP + t];
  for (int t = lane; t < 16 * D; t += 32) {
    int e = t >> 7, d = t & 127;
    // 2*pi / period[d], period = logspace(1e-5, 1, 128)
    float fr = 6.28318530718f * __expf(2.30258509299f * (5.f - 5.f * (float)d * (1.f / 127.f)));
    xf[t] = __sinf(fr * w[eb + e]);
  }
  lds_fence();

  // ---- encoder: h = gelu(p @ W_in1 + b_in1), [16 x 512] ----
  {
    v16bf a = loadA_f32(pb, DP, 0, lane);  // K=32, single K-tile
    for (int nt = 0; nt < 32; nt++) {
      v8f c = v8f_zero();
      c = wmma_bf16(a, loadB(Wf, F_WIN1 + nt, lane), c);
      float bn = Bf[BI_IN1 + nt * 16 + n];
#pragma unroll
      for (int r = 0; r < 8; r++)
        hb[(rb + r) * 512 + nt * 16 + n] = (__bf16)gelu_f(c[r] + bn);
    }
  }
  lds_fence();

  // ---- enc_p = h @ W_in2 + b_in2 -> nkv[e][kv*128+d] (raw) ----
  {
    v16bf ah[16];
#pragma unroll
    for (int kt = 0; kt < 16; kt++) ah[kt] = loadA_bf16(hb, 512, kt * 32, lane);
    for (int nt = 0; nt < 128; nt++) {
      v8f c = v8f_zero();
#pragma unroll
      for (int kt = 0; kt < 16; kt++)
        c = wmma_bf16(ah[kt], loadB(Wf, F_WIN2 + kt * 128 + nt, lane), c);
      int col = nt * 16 + n;
      float bn = Bf[BI_IN2 + col];
#pragma unroll
      for (int r = 0; r < 8; r++)
        nkv[(size_t)(rb + r) * 2048 + col] = (__bf16)(c[r] + bn);
    }
  }
  lds_fence();

  // ---- normalize enc_p once (layer-invariant part of the kv layernorm) ----
  for (int row = lane; row < 256; row += 32) {  // row = e*16 + kv token
    v8bf* rp = (v8bf*)(nkv + row * 128);
    float s = 0.f, s2 = 0.f;
    for (int c8 = 0; c8 < 16; c8++) {
      v8bf t = rp[c8];
#pragma unroll
      for (int i = 0; i < 8; i++) { float f = (float)t[i]; s += f; s2 += f * f; }
    }
    float mu = s * (1.f / 128.f);
    float rs = rsqrtf(s2 * (1.f / 128.f) - mu * mu + 1e-6f);
    for (int c8 = 0; c8 < 16; c8++) {
      v8bf t = rp[c8];
#pragma unroll
      for (int i = 0; i < 8; i++) t[i] = (__bf16)(((float)t[i] - mu) * rs);
      rp[c8] = t;
    }
  }
  lds_fence();

  // ---- transformer layers ----
  for (int l = 0; l < NL; l++) {
    const int fb = F_LAYER + l * F_PER_LAYER;
    const int bbase = BI_LAYER + l * BI_PER_LAYER;
    if (l + 1 < NL)
      __builtin_prefetch(Wf + (size_t)(F_LAYER + (l + 1) * F_PER_LAYER) * FH, 0, 2);

    // q = x @ Wq + bq  (A-fragments hoisted across nt)
    {
      v16bf aq[4];
#pragma unroll
      for (int kt = 0; kt < 4; kt++) aq[kt] = loadA_f32(xf, 128, kt * 32, lane);
      for (int nt = 0; nt < 8; nt++) {
        v8f c = v8f_zero();
#pragma unroll
        for (int kt = 0; kt < 4; kt++)
          c = wmma_bf16(aq[kt], loadB(Wf, fb + FL_Q + kt * 8 + nt, lane), c);
        int col = nt * 16 + n;
        float bn = Bf[bbase + BL_Q + col];
#pragma unroll
        for (int r = 0; r < 8; r++) qf[(rb + r) * 128 + col] = (__bf16)(c[r] + bn);
      }
    }
    lds_fence();

    // per-element KV projections + attention (ln scale folded into Wk'/Wv')
    for (int e = 0; e < 16; e++) {
      const __bf16* ne = nkv + (size_t)e * 2048;  // [16 tok][128]
      v16bf akv[4];
#pragma unroll
      for (int kt = 0; kt < 4; kt++) akv[kt] = loadA_bf16(ne, 128, kt * 32, lane);
      for (int nt = 0; nt < 8; nt++) {
        v8f ck = v8f_zero(), cv = v8f_zero();
#pragma unroll
        for (int kt = 0; kt < 4; kt++) {
          ck = wmma_bf16(akv[kt], loadB(Wf, fb + FL_K + kt * 8 + nt, lane), ck);
          cv = wmma_bf16(akv[kt], loadB(Wf, fb + FL_V + kt * 8 + nt, lane), cv);
        }
        int col = nt * 16 + n;
        float bk_ = Bf[bbase + BL_K + col];
        float bv_ = Bf[bbase + BL_V + col];
#pragma unroll
        for (int r = 0; r < 8; r++) {
          kf[(rb + r) * 128 + col] = (__bf16)(ck[r] + bk_);
          vf[(rb + r) * 128 + col] = (__bf16)(cv[r] + bv_);
        }
      }
      lds_fence();

      // attention: 4 lanes per head (h = lane/4), tokens split by s4 = lane%4
      int h = lane >> 2, s4 = lane & 3;
      float qh[16];
      {
        const v8bf* qp = (const v8bf*)(qf + e * 128 + h * 16);
        v8bf q0 = qp[0], q1 = qp[1];
#pragma unroll
        for (int i = 0; i < 8; i++) { qh[i] = (float)q0[i]; qh[8 + i] = (float)q1[i]; }
      }
      float sc[4];
      float mx = -3.0e38f;
#pragma unroll
      for (int j = 0; j < 4; j++) {
        int tok = s4 * 4 + j;
        const v8bf* kp = (const v8bf*)(kf + tok * 128 + h * 16);
        v8bf k0 = kp[0], k1 = kp[1];
        float dacc = 0.f;
#pragma unroll
        for (int i = 0; i < 8; i++)
          dacc += qh[i] * (float)k0[i] + qh[8 + i] * (float)k1[i];
        sc[j] = dacc * 0.25f;  // 1/sqrt(HD=16)
        mx = fmaxf(mx, sc[j]);
      }
      mx = fmaxf(mx, __shfl_xor(mx, 1));
      mx = fmaxf(mx, __shfl_xor(mx, 2));
      float den = 0.f;
#pragma unroll
      for (int j = 0; j < 4; j++) { sc[j] = __expf(sc[j] - mx); den += sc[j]; }
      den += __shfl_xor(den, 1);
      den += __shfl_xor(den, 2);
      float inv = __builtin_amdgcn_rcpf(den);
      float oa[16];
#pragma unroll
      for (int i = 0; i < 16; i++) oa[i] = 0.f;
#pragma unroll
      for (int j = 0; j < 4; j++) {
        int tok = s4 * 4 + j;
        float aj = sc[j] * inv;
        const v8bf* vp = (const v8bf*)(vf + tok * 128 + h * 16);
        v8bf v0 = vp[0], v1 = vp[1];
#pragma unroll
        for (int i = 0; i < 8; i++) {
          oa[i]     += aj * (float)v0[i];
          oa[8 + i] += aj * (float)v1[i];
        }
      }
#pragma unroll
      for (int i = 0; i < 16; i++) {
        oa[i] += __shfl_xor(oa[i], 1);
        oa[i] += __shfl_xor(oa[i], 2);
      }
      __bf16* oe = ob + e * 128;
#pragma unroll
      for (int t = 0; t < 4; t++)
        oe[h * 16 + s4 * 4 + t] = (__bf16)oa[s4 * 4 + t];
      lds_fence();
    }

    // x += o @ Wo + bo
    {
      v16bf ao[4];
#pragma unroll
      for (int kt = 0; kt < 4; kt++) ao[kt] = loadA_bf16(ob, 128, kt * 32, lane);
      for (int nt = 0; nt < 8; nt++) {
        v8f c = v8f_zero();
#pragma unroll
        for (int kt = 0; kt < 4; kt++)
          c = wmma_bf16(ao[kt], loadB(Wf, fb + FL_O + kt * 8 + nt, lane), c);
        int col = nt * 16 + n;
        float bn = Bf[bbase + BL_O + col];
#pragma unroll
        for (int r = 0; r < 8; r++) xf[(rb + r) * 128 + col] += c[r] + bn;
      }
    }
    lds_fence();

    // hh = normalize(x)  (lxs/lxb folded into W_m1'/bm1')
    {
      int row = lane & 15;  // lanes 16..31 duplicate; harmless
      float s = 0.f, s2 = 0.f;
      for (int d = 0; d < 128; d++) { float v = xf[row * 128 + d]; s += v; s2 += v * v; }
      float mu = s * (1.f / 128.f);
      float rs = rsqrtf(s2 * (1.f / 128.f) - mu * mu + 1e-6f);
      if (lane < 16)
        for (int d = 0; d < 128; d++)
          hh[row * 128 + d] = (__bf16)((xf[row * 128 + d] - mu) * rs);
    }
    lds_fence();

    // t = gelu(hh @ W_m1' + bm1'), [16 x 256]
    {
      v16bf am[4];
#pragma unroll
      for (int kt = 0; kt < 4; kt++) am[kt] = loadA_bf16(hh, 128, kt * 32, lane);
      for (int nt = 0; nt < 16; nt++) {
        v8f c = v8f_zero();
#pragma unroll
        for (int kt = 0; kt < 4; kt++)
          c = wmma_bf16(am[kt], loadB(Wf, fb + FL_M1 + kt * 16 + nt, lane), c);
        int col = nt * 16 + n;
        float bn = Bf[bbase + BL_M1 + col];
#pragma unroll
        for (int r = 0; r < 8; r++)
          tb[(rb + r) * 256 + col] = (__bf16)gelu_f(c[r] + bn);
      }
    }
    lds_fence();

    // x += t @ W_m2 + bm2
    {
      v16bf am[8];
#pragma unroll
      for (int kt = 0; kt < 8; kt++) am[kt] = loadA_bf16(tb, 256, kt * 32, lane);
      for (int nt = 0; nt < 8; nt++) {
        v8f c = v8f_zero();
#pragma unroll
        for (int kt = 0; kt < 8; kt++)
          c = wmma_bf16(am[kt], loadB(Wf, fb + FL_M2 + kt * 8 + nt, lane), c);
        int col = nt * 16 + n;
        float bn = Bf[bbase + BL_M2 + col];
#pragma unroll
        for (int r = 0; r < 8; r++) xf[(rb + r) * 128 + col] += c[r] + bn;
      }
    }
    lds_fence();
  }

  // ---- head: gelu(x@W_o1+b1) -> gelu(@W_o2+b2) -> @W_o3+b3 ----
  {
    v16bf a4[4];
#pragma unroll
    for (int kt = 0; kt < 4; kt++) a4[kt] = loadA_f32(xf, 128, kt * 32, lane);
    for (int nt = 0; nt < 16; nt++) {
      v8f c = v8f_zero();
#pragma unroll
      for (int kt = 0; kt < 4; kt++)
        c = wmma_bf16(a4[kt], loadB(Wf, F_WO1 + kt * 16 + nt, lane), c);
      int col = nt * 16 + n;
      float bn = Bf[BI_O1 + col];
#pragma unroll
      for (int r = 0; r < 8; r++)
        ha[(rb + r) * 256 + col] = (__bf16)gelu_f(c[r] + bn);
    }
  }
  lds_fence();
  {
    v16bf a8[8];
#pragma unroll
    for (int kt = 0; kt < 8; kt++) a8[kt] = loadA_bf16(ha, 256, kt * 32, lane);
    for (int nt = 0; nt < 16; nt++) {
      v8f c = v8f_zero();
#pragma unroll
      for (int kt = 0; kt < 8; kt++)
        c = wmma_bf16(a8[kt], loadB(Wf, F_WO2 + kt * 16 + nt, lane), c);
      int col = nt * 16 + n;
      float bn = Bf[BI_O2 + col];
#pragma unroll
      for (int r = 0; r < 8; r++)
        hc[(rb + r) * 256 + col] = (__bf16)gelu_f(c[r] + bn);
    }
  }
  lds_fence();
  {
    v8f c = v8f_zero();
#pragma unroll
    for (int kt = 0; kt < 8; kt++)
      c = wmma_bf16(loadA_bf16(hc, 256, kt * 32, lane),
                    loadB(Wf, F_WO3 + kt, lane), c);
    if (n < 2) {
      float bn = Bf[BI_O3 + n];
#pragma unroll
      for (int r = 0; r < 8; r++)
        out[(size_t)(eb + rb + r) * 2 + n] = c[r] + bn;
    }
  }
}

// ---------------------------------------------------------------------------
// host
// ---------------------------------------------------------------------------
extern "C" void kernel_launch(void* const* d_in, const int* in_sizes, int n_in,
                              void* d_out, int out_size, void* d_ws, size_t ws_size,
                              hipStream_t stream) {
  (void)in_sizes; (void)n_in; (void)out_size;
  if (ws_size < WS_TOTAL) return;

  const float* p     = (const float*)d_in[0];
  const float* w     = (const float*)d_in[1];
  const float* W_in1 = (const float*)d_in[2];
  const float* b_in1 = (const float*)d_in[3];
  const float* W_in2 = (const float*)d_in[4];
  const float* b_in2 = (const float*)d_in[5];
  const float* lks   = (const float*)d_in[6];
  const float* lkb   = (const float*)d_in[7];
  const float* Wq    = (const float*)d_in[8];
  const float* bq    = (const float*)d_in[9];
  const float* Wk    = (const float*)d_in[10];
  const float* bk    = (const float*)d_in[11];
  const float* Wv    = (const float*)d_in[12];
  const float* bv    = (const float*)d_in[13];
  const float* Wo    = (const float*)d_in[14];
  const float* bo    = (const float*)d_in[15];
  const float* lxs   = (const float*)d_in[16];
  const float* lxb   = (const float*)d_in[17];
  const float* Wm1   = (const float*)d_in[18];
  const float* bm1   = (const float*)d_in[19];
  const float* Wm2   = (const float*)d_in[20];
  const float* bm2   = (const float*)d_in[21];
  const float* Wo1   = (const float*)d_in[22];
  const float* bo1   = (const float*)d_in[23];
  const float* Wo2   = (const float*)d_in[24];
  const float* bo2   = (const float*)d_in[25];
  const float* Wo3   = (const float*)d_in[26];
  const float* bo3   = (const float*)d_in[27];

  unsigned short* wsW = (unsigned short*)d_ws;
  float* wsB = (float*)((char*)d_ws + WBYTES);

  // --- pack weights into bf16 B-fragments (fold layernorm scales) ---
  pack_w<<<32,   32, 0, stream>>>(W_in1, nullptr, 32, 512, 32,  wsW + (size_t)F_WIN1 * FH);
  pack_w<<<2048, 32, 0, stream>>>(W_in2, nullptr, 512, 2048, 128, wsW + (size_t)F_WIN2 * FH);
  for (int l = 0; l < NL; l++) {
    size_t fbase = (size_t)(F_LAYER + l * F_PER_LAYER) * FH;
    pack_w<<<32, 32, 0, stream>>>(Wq + (size_t)l * D * D, nullptr,        128, 128, 8,  wsW + fbase + (size_t)FL_Q  * FH);
    pack_w<<<32, 32, 0, stream>>>(Wk + (size_t)l * D * D, lks + l * D,    128, 128, 8,  wsW + fbase + (size_t)FL_K  * FH);
    pack_w<<<32, 32, 0, stream>>>(Wv + (size_t)l * D * D, lks + l * D,    128, 128, 8,  wsW + fbase + (size_t)FL_V  * FH);
    pack_w<<<32, 32, 0, stream>>>(Wo + (size_t)l * D * D, nullptr,        128, 128, 8,  wsW + fbase + (size_t)FL_O  * FH);
    pack_w<<<64, 32, 0, stream>>>(Wm1 + (size_t)l * D * 2 * D, lxs + l * D, 128, 256, 16, wsW + fbase + (size_t)FL_M1 * FH);
    pack_w<<<64, 32, 0, stream>>>(Wm2 + (size_t)l * 2 * D * D, nullptr,     256, 128, 8,  wsW + fbase + (size_t)FL_M2 * FH);
  }
  pack_w<<<64,  32, 0, stream>>>(Wo1, nullptr, 128, 256, 16, wsW + (size_t)F_WO1 * FH);
  pack_w<<<128, 32, 0, stream>>>(Wo2, nullptr, 256, 256, 16, wsW + (size_t)F_WO2 * FH);
  pack_w<<<8,   32, 0, stream>>>(Wo3, nullptr, 256, 2,   1,  wsW + (size_t)F_WO3 * FH);

  // --- effective biases (fold layernorm bias through k/v/m1 weights) ---
  fold_bias<<<8,  64, 0, stream>>>(b_in1, nullptr, nullptr, 0, 512,  wsB + BI_IN1);
  fold_bias<<<32, 64, 0, stream>>>(b_in2, nullptr, nullptr, 0, 2048, wsB + BI_IN2);
  for (int l = 0; l < NL; l++) {
    float* bl = wsB + BI_LAYER + l * BI_PER_LAYER;
    fold_bias<<<2, 64, 0, stream>>>(bq + l * D,  nullptr,     nullptr,                128, 128, bl + BL_Q);
    fold_bias<<<2, 64, 0, stream>>>(bk + l * D,  lkb + l * D, Wk + (size_t)l * D * D, 128, 128, bl + BL_K);
    fold_bias<<<2, 64, 0, stream>>>(bv + l * D,  lkb + l * D, Wv + (size_t)l * D * D, 128, 128, bl + BL_V);
    fold_bias<<<2, 64, 0, stream>>>(bo + l * D,  nullptr,     nullptr,                128, 128, bl + BL_O);
    fold_bias<<<4, 64, 0, stream>>>(bm1 + l * 2 * D, lxb + l * D, Wm1 + (size_t)l * D * 2 * D, 128, 256, bl + BL_M1);
    fold_bias<<<2, 64, 0, stream>>>(bm2 + l * D, nullptr,     nullptr,                256, 128, bl + BL_M2);
  }
  fold_bias<<<4, 64, 0, stream>>>(bo1, nullptr, nullptr, 0, 256, wsB + BI_O1);
  fold_bias<<<4, 64, 0, stream>>>(bo2, nullptr, nullptr, 0, 256, wsB + BI_O2);
  fold_bias<<<1, 64, 0, stream>>>(bo3, nullptr, nullptr, 0, 2,   wsB + BI_O3);

  // --- main fused kernel: 1024 waves, 16 batch elements each ---
  mlp_att_kernel<<<BATCH / 16, 32, SMEM_BYTES, stream>>>(p, w, wsW, wsB, (float*)d_out);
}
